// DecoderLayer_90074054132191
// MI455X (gfx1250) — matrix operand order repaired and checked by reading, not compiled, not run on hardware
//
#include <hip/hip_runtime.h>

// ---- problem dims (fixed by the reference) ----
#define BB 2
#define LL 1024
#define DDIM 1024
#define HH 16
#define DHD 64
#define FDIM 4096
#define BLROWS (BB * LL)   // 2048

typedef __bf16 bf16_t;
typedef __attribute__((ext_vector_type(16))) __bf16 v16bf;
typedef __attribute__((ext_vector_type(8)))  __bf16 v8bf;
typedef __attribute__((ext_vector_type(8)))  float  v8f;
typedef __attribute__((ext_vector_type(4)))  unsigned int v4u;

__device__ __forceinline__ bf16_t f2bf(float f) {
  unsigned u = __builtin_bit_cast(unsigned, f);
  unsigned r = u + 0x7FFFu + ((u >> 16) & 1u);   // round-to-nearest-even
  unsigned short s = (unsigned short)(r >> 16);
  return __builtin_bit_cast(bf16_t, s);
}

__device__ __forceinline__ v16bf cat8(v8bf lo, v8bf hi) {
  v16bf r;
#pragma unroll
  for (int i = 0; i < 8; ++i) { r[i] = lo[i]; r[i + 8] = hi[i]; }
  return r;
}

// CDNA5 async global->LDS copy (ASYNCcnt-tracked, 16B per lane).
// lds_addr: raw 32-bit LDS byte address; g: per-lane 64-bit global address.
__device__ __forceinline__ void async_cp_b128(unsigned lds_addr, const void* g) {
  asm volatile("global_load_async_to_lds_b128 %0, %1, off"
               :: "v"(lds_addr), "v"(g) : "memory");
}
__device__ __forceinline__ void wait_asynccnt0() {
  asm volatile("s_wait_asynccnt 0x0" ::: "memory");
}

// ---------------------------------------------------------------------------
// Tiled bf16 WMMA GEMM: C[M,N] = act(A[M,K] @ B[K,N] + bias[N])
// ACT: 0=none, 1=sigmoid for col<act_limit, 2=gelu(exact erf)
// Requires M%128==0, N%128==0, K%32==0 (true for all uses here).
// A tile is staged via GLOBAL_LOAD_ASYNC_TO_LDS_B128 (one barrier / k-step);
// B tile is staged through registers with an LDS transpose to [n][k] so both
// WMMA fragments are contiguous ds_load_b128 pairs.
// ---------------------------------------------------------------------------
template <int ACT, bool OUT_BF16>
__global__ __launch_bounds__(256)
void gemm_bf16_wmma(const bf16_t* __restrict__ A, const bf16_t* __restrict__ B,
                    const float* __restrict__ bias, void* __restrict__ Cout,
                    int M, int N, int K, int act_limit) {
  constexpr int BM = 128, BN = 128, BK = 32;
  __shared__ alignas(32) bf16_t Alds[2][BM * BK];      // [m][k]
  __shared__ alignas(32) bf16_t Blds[2][BN * BK];      // transposed: [n][k]

  const int tid  = threadIdx.x;
  const int lane = tid & 31;
  const int wave = tid >> 5;
  const int wr = (wave >> 2) * 64;   // wave row offset within block tile
  const int wc = (wave & 3) * 32;    // wave col offset
  const int bm = blockIdx.y * BM;
  const int bn = blockIdx.x * BN;

  const int kh = lane >> 4;          // lane half selects K sub-range
  const int mn = lane & 15;          // m (A) / n (B,D) index within 16

  v8f acc[4][2];
#pragma unroll
  for (int i = 0; i < 4; ++i)
#pragma unroll
    for (int j = 0; j < 2; ++j) acc[i][j] = v8f{};

  // staging indices
  const int arow = tid >> 1;           // 0..127
  const int aseg = (tid & 1) * 16;     // 0/16 within BK
  const int brow = tid >> 3;           // 0..31  (k within tile)
  const int bseg = (tid & 7) * 16;     // n segment within BN

  // raw LDS byte address for this thread's A staging slot (aperture low bits)
  const unsigned a_lds = (unsigned)(size_t)(&Alds[0][0]) +
                         (unsigned)(arow * BK + aseg) * 2u;

  v4u breg[2];
  auto asyncA = [&](int buf, int kt) {
    const bf16_t* ap = A + (size_t)(bm + arow) * K + kt * BK + aseg;
    const unsigned dst = a_lds + (unsigned)buf * (BM * BK * 2);
    async_cp_b128(dst, ap);
    async_cp_b128(dst + 16, ap + 8);
  };
  auto gloadB = [&](int kt) {
    const bf16_t* bp = B + (size_t)(kt * BK + brow) * N + bn + bseg;
    breg[0] = *(const v4u*)bp;
    breg[1] = *(const v4u*)(bp + 8);
  };
  auto sstoreB = [&](int buf) {
    const bf16_t* br = (const bf16_t*)&breg[0];
#pragma unroll
    for (int i = 0; i < 16; ++i)
      Blds[buf][(bseg + i) * BK + brow] = br[i];   // transpose into [n][k]
  };

  const int nk = K / BK;
  asyncA(0, 0);
  gloadB(0);
  sstoreB(0);
  wait_asynccnt0();
  __syncthreads();

  for (int kt = 0; kt < nk; ++kt) {
    if (kt + 1 < nk) {
      asyncA((kt + 1) & 1, kt + 1);   // async copy overlaps WMMA below
      gloadB(kt + 1);
    }
    const int buf = kt & 1;

    v16bf afrag[4], bfrag[2];
#pragma unroll
    for (int tm = 0; tm < 4; ++tm) {
      const int m = wr + tm * 16 + mn;
      v8bf lo = *(const v8bf*)&Alds[buf][m * BK + 8 * kh];
      v8bf hi = *(const v8bf*)&Alds[buf][m * BK + 16 + 8 * kh];
      afrag[tm] = cat8(lo, hi);       // per-lane K order: 8kh..+7, 16+8kh..+7
    }
#pragma unroll
    for (int tn = 0; tn < 2; ++tn) {
      const int n = wc + tn * 16 + mn;
      v8bf blo = *(const v8bf*)&Blds[buf][n * BK + 8 * kh];
      v8bf bhi = *(const v8bf*)&Blds[buf][n * BK + 16 + 8 * kh];
      bfrag[tn] = cat8(blo, bhi); // K order: 8kh..+7, 16+8kh..+7
    }

#pragma unroll
    for (int tm = 0; tm < 4; ++tm)
#pragma unroll
      for (int tn = 0; tn < 2; ++tn)
        acc[tm][tn] = __builtin_amdgcn_wmma_f32_16x16x32_bf16(
            false, afrag[tm], false, bfrag[tn], (short)0, acc[tm][tn],
            false, false);

    if (kt + 1 < nk) {
      sstoreB((kt + 1) & 1);   // other-parity buffer: no live readers
      wait_asynccnt0();        // A tile for kt+1 landed in LDS
      __syncthreads();         // publish both tiles (single barrier / iter)
    }
  }

  // epilogue; C/D layout: lane half kh -> m = v + 8*kh, n = lane&15
  float*  Cf = (float*)Cout;
  bf16_t* Cb = (bf16_t*)Cout;
#pragma unroll
  for (int tm = 0; tm < 4; ++tm) {
#pragma unroll
    for (int tn = 0; tn < 2; ++tn) {
      const int col = bn + wc + tn * 16 + mn;
      const float bv = bias ? bias[col] : 0.f;
#pragma unroll
      for (int v = 0; v < 8; ++v) {
        const int row = bm + wr + tm * 16 + v + 8 * kh;
        float x = acc[tm][tn][v] + bv;
        if (ACT == 1) {
          if (col < act_limit) x = 1.f / (1.f + __expf(-x));
        } else if (ACT == 2) {
          x = 0.5f * x * (1.f + erff(x * 0.70710678118654752f));
        }
        if (OUT_BF16) Cb[(size_t)row * N + col] = f2bf(x);
        else          Cf[(size_t)row * N + col] = x;
      }
    }
  }
}

// ---------------------------------------------------------------------------
// Causal linear-attention scan with gating. One wave per (b,h).
// qkv: [BL][3*D] f32, sigmoid already applied to q,k by GEMM epilogue.
// attn_out: [BL][D] bf16 in (b, l, h*DH + c) layout (heads re-interleaved).
// Lane owns columns c0=2*lane, c1=2*lane+1 of the 64x64 kv state (registers)
// and d-components d0,d1 of the gating cumsums; gating dots via shfl_xor.
// ---------------------------------------------------------------------------
__global__ __launch_bounds__(32)
void attn_scan(const float* __restrict__ qkv, bf16_t* __restrict__ attn_out) {
  const float EPS = 1e-6f;
  const int bh = blockIdx.x;
  const int b  = bh / HH;
  const int h  = bh % HH;
  const int lane = threadIdx.x;
  const int d0 = 2 * lane, d1 = 2 * lane + 1;

  float st0[DHD], st1[DHD];
#pragma unroll
  for (int r = 0; r < DHD; ++r) { st0[r] = 0.f; st1[r] = 0.f; }

  float csk0 = 0.f, csk1 = 0.f, csq0 = 0.f, csq1 = 0.f;
  float cso0 = 0.f, cso1 = 0.f, csi0 = 0.f, csi1 = 0.f;  // cumsum k*src_out, q*sink_in
  float cume = 0.f;

  __shared__ float kbuf[DHD];
  __shared__ float qsbuf[DHD];

  for (int l = 0; l < LL; ++l) {
    const size_t row = (size_t)(b * LL + l) * (3 * DDIM) + h * DHD;
    const float q0 = qkv[row + d0],              q1 = qkv[row + d1];
    const float k0 = qkv[row + DDIM + d0],       k1 = qkv[row + DDIM + d1];
    const float v0 = qkv[row + 2 * DDIM + d0],   v1 = qkv[row + 2 * DDIM + d1];

    csk0 += k0; csk1 += k1; csq0 += q0; csq1 += q1;

    float p1 = (q0 + EPS) * (csk0 + EPS) + (q1 + EPS) * (csk1 + EPS);
    float p2 = (k0 + EPS) * (csq0 + EPS) + (k1 + EPS) * (csq1 + EPS);
#pragma unroll
    for (int off = 16; off; off >>= 1) {
      p1 += __shfl_xor(p1, off, 32);
      p2 += __shfl_xor(p2, off, 32);
    }
    const float normal  = (float)(l + 1);
    const float sink_in = normal / p1;
    const float src_out = normal / p2;

    cso0 += k0 * src_out; cso1 += k1 * src_out;
    csi0 += q0 * sink_in; csi1 += q1 * sink_in;

    float p3 = (q0 + EPS) * (cso0 + EPS) + (q1 + EPS) * (cso1 + EPS);
    float p4 = (k0 + EPS) * (csi0 + EPS) + (k1 + EPS) * (csi1 + EPS);
#pragma unroll
    for (int off = 16; off; off >>= 1) {
      p3 += __shfl_xor(p3, off, 32);
      p4 += __shfl_xor(p4, off, 32);
    }
    const float cons_sink  = p3 / normal;
    float cons_src = p4 / normal;
    cons_src = fminf(fmaxf(cons_src, -1.f), 1.f);
    const float sink_alloc = 1.f / (1.f + __expf(-cons_sink));
    const float e = __expf(cons_src);
    cume += e;
    const float src_comp = e / cume * normal;

    const float qsc = sink_in / normal;
    const float vs0 = v0 * src_comp, vs1 = v1 * src_comp;

    kbuf[d0] = k0;        kbuf[d1] = k1;
    qsbuf[d0] = q0 * qsc; qsbuf[d1] = q1 * qsc;
    __syncthreads();

    float acc0 = 0.f, acc1 = 0.f;
#pragma unroll
    for (int r = 0; r < DHD; ++r) {
      const float kr = kbuf[r], qr = qsbuf[r];
      st0[r] += kr * vs0; acc0 += qr * st0[r];
      st1[r] += kr * vs1; acc1 += qr * st1[r];
    }
    __syncthreads();

    const size_t orow = (size_t)(b * LL + l) * DDIM + h * DHD;
    attn_out[orow + d0] = f2bf(acc0 * sink_alloc);
    attn_out[orow + d1] = f2bf(acc1 * sink_alloc);
  }
}

// ---------------------------------------------------------------------------
// out = LayerNorm(x + y) * g + b  (eps=1e-5). Optionally also emit bf16 copy.
// One 256-thread block per row, D=1024 (4 elements / thread).
// ---------------------------------------------------------------------------
__global__ __launch_bounds__(256)
void add_layernorm(const float* __restrict__ x, const float* __restrict__ y,
                   const float* __restrict__ g, const float* __restrict__ b,
                   float* __restrict__ outf, bf16_t* __restrict__ outb) {
  const int row = blockIdx.x;
  const int tid = threadIdx.x;
  const int lane = tid & 31, wave = tid >> 5;
  __shared__ float sm[8];

  const float* xr = x + (size_t)row * DDIM;
  const float* yr = y + (size_t)row * DDIM;

  float vals[4];
  float s = 0.f;
#pragma unroll
  for (int i = 0; i < 4; ++i) {
    const int idx = tid + i * 256;
    vals[i] = xr[idx] + yr[idx];
    s += vals[i];
  }
#pragma unroll
  for (int off = 16; off; off >>= 1) s += __shfl_xor(s, off, 32);
  if (lane == 0) sm[wave] = s;
  __syncthreads();
  float tot = 0.f;
#pragma unroll
  for (int w = 0; w < 8; ++w) tot += sm[w];
  const float mean = tot * (1.f / DDIM);

  float vsum = 0.f;
#pragma unroll
  for (int i = 0; i < 4; ++i) {
    const float d = vals[i] - mean;
    vsum += d * d;
  }
#pragma unroll
  for (int off = 16; off; off >>= 1) vsum += __shfl_xor(vsum, off, 32);
  __syncthreads();
  if (lane == 0) sm[wave] = vsum;
  __syncthreads();
  float vtot = 0.f;
#pragma unroll
  for (int w = 0; w < 8; ++w) vtot += sm[w];
  const float rstd = rsqrtf(vtot * (1.f / DDIM) + 1e-5f);

#pragma unroll
  for (int i = 0; i < 4; ++i) {
    const int idx = tid + i * 256;
    const float o = (vals[i] - mean) * rstd * g[idx] + b[idx];
    outf[(size_t)row * DDIM + idx] = o;
    if (outb) outb[(size_t)row * DDIM + idx] = f2bf(o);
  }
}

// ---------------------------------------------------------------------------
// helpers: f32 -> bf16 cast; QKV weight/bias packing into [D, 3D]
// ---------------------------------------------------------------------------
__global__ void cast_to_bf16(const float* __restrict__ in,
                             bf16_t* __restrict__ out, int n) {
  const int i = blockIdx.x * 256 + threadIdx.x;
  if (i < n) out[i] = f2bf(in[i]);
}

__global__ void pack_qkv_w(const float* __restrict__ wq,
                           const float* __restrict__ wk,
                           const float* __restrict__ wv,
                           bf16_t* __restrict__ w) {
  const int i = blockIdx.x * 256 + threadIdx.x;
  if (i >= DDIM * DDIM) return;
  const int k = i / DDIM, n = i % DDIM;
  const size_t base = (size_t)k * (3 * DDIM);
  w[base + n]            = f2bf(wq[i]);
  w[base + DDIM + n]     = f2bf(wk[i]);
  w[base + 2 * DDIM + n] = f2bf(wv[i]);
}

__global__ void pack_qkv_b(const float* __restrict__ bq,
                           const float* __restrict__ bk,
                           const float* __restrict__ bv,
                           float* __restrict__ bo) {
  const int i = blockIdx.x * 256 + threadIdx.x;
  if (i >= DDIM) return;
  bo[i] = bq[i]; bo[DDIM + i] = bk[i]; bo[2 * DDIM + i] = bv[i];
}

// ---------------------------------------------------------------------------
extern "C" void kernel_launch(void* const* d_in, const int* in_sizes, int n_in,
                              void* d_out, int out_size, void* d_ws, size_t ws_size,
                              hipStream_t stream) {
  const float* inp  = (const float*)d_in[0];
  const float* Wq   = (const float*)d_in[2];
  const float* bq   = (const float*)d_in[3];
  const float* Wk   = (const float*)d_in[4];
  const float* bk   = (const float*)d_in[5];
  const float* Wv   = (const float*)d_in[6];
  const float* bv   = (const float*)d_in[7];
  const float* Wo   = (const float*)d_in[8];
  const float* bo   = (const float*)d_in[9];
  const float* ln1g = (const float*)d_in[10];
  const float* ln1b = (const float*)d_in[11];
  const float* W1   = (const float*)d_in[12];
  const float* b1   = (const float*)d_in[13];
  const float* W2   = (const float*)d_in[14];
  const float* b2   = (const float*)d_in[15];
  const float* ln2g = (const float*)d_in[16];
  const float* ln2b = (const float*)d_in[17];
  float* out = (float*)d_out;

  // workspace bump allocator (256B aligned)
  char* ws = (char*)d_ws;
  size_t off = 0;
  auto alloc = [&](size_t bytes) -> void* {
    void* p = ws + off;
    off += (bytes + 255) & ~(size_t)255;
    return p;
  };
  bf16_t* xbf     = (bf16_t*)alloc((size_t)BLROWS * DDIM * 2);
  bf16_t* wqkv    = (bf16_t*)alloc((size_t)DDIM * 3 * DDIM * 2);
  float*  bqkv    = (float*) alloc((size_t)3 * DDIM * 4);
  bf16_t* wobf    = (bf16_t*)alloc((size_t)DDIM * DDIM * 2);
  bf16_t* w1bf    = (bf16_t*)alloc((size_t)DDIM * FDIM * 2);
  bf16_t* w2bf    = (bf16_t*)alloc((size_t)FDIM * DDIM * 2);
  float*  qkv     = (float*) alloc((size_t)BLROWS * 3 * DDIM * 4);
  bf16_t* attnb   = (bf16_t*)alloc((size_t)BLROWS * DDIM * 2);
  float*  attnprj = (float*) alloc((size_t)BLROWS * DDIM * 4);
  float*  hf      = (float*) alloc((size_t)BLROWS * DDIM * 4);
  bf16_t* hb      = (bf16_t*)alloc((size_t)BLROWS * DDIM * 2);
  bf16_t* ffn1    = (bf16_t*)alloc((size_t)BLROWS * FDIM * 2);
  float*  ffn2    = (float*) alloc((size_t)BLROWS * DDIM * 4);
  (void)ws_size; (void)in_sizes; (void)n_in; (void)out_size;

  // 1) casts / packing
  {
    int n = BLROWS * DDIM;
    cast_to_bf16<<<(n + 255) / 256, 256, 0, stream>>>(inp, xbf, n);
    pack_qkv_w<<<(DDIM * DDIM + 255) / 256, 256, 0, stream>>>(Wq, Wk, Wv, wqkv);
    pack_qkv_b<<<(DDIM + 255) / 256, 256, 0, stream>>>(bq, bk, bv, bqkv);
    n = DDIM * DDIM;
    cast_to_bf16<<<(n + 255) / 256, 256, 0, stream>>>(Wo, wobf, n);
    n = DDIM * FDIM;
    cast_to_bf16<<<(n + 255) / 256, 256, 0, stream>>>(W1, w1bf, n);
    cast_to_bf16<<<(n + 255) / 256, 256, 0, stream>>>(W2, w2bf, n);
  }

  // 2) QKV = x @ [Wq|Wk|Wv] + b ; sigmoid on q,k columns (col < 2D)
  {
    dim3 grid(3 * DDIM / 128, BLROWS / 128);
    gemm_bf16_wmma<1, false><<<grid, 256, 0, stream>>>(
        xbf, wqkv, bqkv, qkv, BLROWS, 3 * DDIM, DDIM, 2 * DDIM);
  }

  // 3) gated causal linear-attention scan
  attn_scan<<<BB * HH, 32, 0, stream>>>(qkv, attnb);

  // 4) attn @ Wo + bo
  {
    dim3 grid(DDIM / 128, BLROWS / 128);
    gemm_bf16_wmma<0, false><<<grid, 256, 0, stream>>>(
        attnb, wobf, bo, attnprj, BLROWS, DDIM, DDIM, 0);
  }

  // 5) h = LN1(inputs + attn)
  add_layernorm<<<BLROWS, 256, 0, stream>>>(inp, attnprj, ln1g, ln1b, hf, hb);

  // 6) ffn1 = gelu(h @ W1 + b1), emitted as bf16
  {
    dim3 grid(FDIM / 128, BLROWS / 128);
    gemm_bf16_wmma<2, true><<<grid, 256, 0, stream>>>(
        hb, w1bf, b1, ffn1, BLROWS, FDIM, DDIM, 0);
  }

  // 7) ffn2 = ffn1 @ W2 + b2
  {
    dim3 grid(DDIM / 128, BLROWS / 128);
    gemm_bf16_wmma<0, false><<<grid, 256, 0, stream>>>(
        ffn1, w2bf, b2, ffn2, BLROWS, DDIM, FDIM, 0);
  }

  // 8) out = LN2(h + ffn2)
  add_layernorm<<<BLROWS, 256, 0, stream>>>(hf, ffn2, ln2g, ln2b, out, nullptr);
}
